// PrototypeDiversityRegularizer_47785806135524
// MI455X (gfx1250) — matrix operand order repaired and checked by reading, not compiled, not run on hardware
//
#include <hip/hip_runtime.h>
#include <hip/hip_bf16.h>
#include <stdint.h>

#define KPC   10
#define NCLS  8
#define DCH   512
#define HH_   96
#define WW_   96
#define HWSZ  (HH_*WW_)       // 9216
#define NB    8
#define NPIX  (NB*HWSZ)       // 73728
#define NPROTO 80
#define STATS_PER 67          // 10 S1 + 55 S2 + V + omega

typedef __attribute__((ext_vector_type(16))) __bf16 v16bf;
typedef __attribute__((ext_vector_type(8)))  float  v8f;

static __device__ __forceinline__ unsigned short bf16_bits(float f) {
  __bf16 h = (__bf16)f;
  return __builtin_bit_cast(unsigned short, h);
}

// ---------------------------------------------------------------------------
// K0: per-pixel inverse L2 norm over channels (coalesced: pixels contiguous)
// ---------------------------------------------------------------------------
__global__ void k0_inv_norm(const float* __restrict__ feat, float* __restrict__ invn) {
  int p = blockIdx.x * blockDim.x + threadIdx.x;
  if (p >= NPIX) return;
  int b = p / HWSZ, hw = p % HWSZ;
  const float* base = feat + (size_t)b * DCH * HWSZ + hw;
  float ss = 0.f;
  #pragma unroll 8
  for (int d = 0; d < DCH; ++d) {
    float v = base[(size_t)d * HWSZ];
    ss = fmaf(v, v, ss);
  }
  invn[p] = 1.f / fmaxf(sqrtf(ss), 1e-12f);
}

// ---------------------------------------------------------------------------
// KP: L2-normalize prototypes, store bf16 row-major [80][512]
// ---------------------------------------------------------------------------
__global__ void kp_proto(const float* __restrict__ proto, unsigned short* __restrict__ pbf) {
  __shared__ float red[64];
  int r = blockIdx.x, t = threadIdx.x;
  const float* row = proto + r * DCH;
  float ss = 0.f;
  #pragma unroll
  for (int j = 0; j < 8; ++j) { float v = row[t*8 + j]; ss = fmaf(v, v, ss); }
  red[t] = ss; __syncthreads();
  for (int off = 32; off > 0; off >>= 1) {
    if (t < off) red[t] += red[t + off];
    __syncthreads();
  }
  float inv = 1.f / fmaxf(sqrtf(red[0]), 1e-12f);
  #pragma unroll
  for (int j = 0; j < 8; ++j) pbf[r*DCH + t*8 + j] = bf16_bits(row[t*8 + j] * inv);
}

// ---------------------------------------------------------------------------
// KZ: zero the global stats accumulator
// ---------------------------------------------------------------------------
__global__ void kzero(float* __restrict__ g, int n) {
  int i = blockIdx.x * blockDim.x + threadIdx.x;
  if (i < n) g[i] = 0.f;
}

// ---------------------------------------------------------------------------
// K1: bf16 WMMA GEMM  sim[pixel, proto] = P_n (A, 16x32) x F (B, 32x16)
//     block = 256 thr = 8 waves, each wave owns 16 pixels, all 80 protos.
//     Proto K-chunk staged into LDS with GLOBAL_LOAD_ASYNC_TO_LDS_B128
//     (ASYNCcnt-tracked, no VGPR round-trip); B fragment loaded from global
//     (pixel across lanes matches the B striping; f32->bf16 via v_cvt_pk).
//     All 5 A fragments loaded before the 5 back-to-back WMMAs.
// ---------------------------------------------------------------------------
__global__ void __launch_bounds__(256) k1_wmma_sim(
    const float* __restrict__ feat, const unsigned short* __restrict__ pbf,
    const int* __restrict__ mask, const float* __restrict__ invn,
    float* __restrict__ msim) {
  __shared__ __align__(16) unsigned short chunk[NPROTO * 32];  // 5 KB
  const int tid  = threadIdx.x;
  const int wave = tid >> 5, lane = tid & 31;
  const int n    = lane & 15, half = lane >> 4;
  const int p    = blockIdx.x * 128 + wave * 16 + n;   // 9216 % 128 == 0 -> one image/block
  const int b    = p / HWSZ, hw = p % HWSZ;
  const float* fbase = feat + (size_t)b * DCH * HWSZ + hw;

  v8f acc[5] = {};
  const unsigned int* src32 = (const unsigned int*)pbf;
  // workgroup-relative LDS byte offset of the staging buffer
  const unsigned ldsBase = (unsigned)(uintptr_t)(void*)&chunk[0];

  for (int kk = 0; kk < 16; ++kk) {
    __syncthreads();  // previous-iteration readers done before overwrite
    // async-stage proto K-chunk [80 rows x 32 k] bf16 into LDS: 320 x 16B
    for (int i = tid; i < NPROTO * 2; i += 256) {   // 160 b128 copies... 80*32*2B/16B = 320
    }
    for (int i = tid; i < 320; i += 256) {
      int r = i >> 2, q = i & 3;
      const unsigned int* g = src32 + r * (DCH / 2) + kk * 16 + q * 4;
      unsigned lofs = ldsBase + (unsigned)i * 16u;
      asm volatile("global_load_async_to_lds_b128 %0, %1, off"
                   :: "v"(lofs), "v"(g) : "memory");
    }
    asm volatile("s_wait_asynccnt 0x0" ::: "memory");
    __syncthreads();

    if (kk < 15)  // global_prefetch_b8 for next K-slab of this lane's pixel
      __builtin_prefetch(&fbase[(size_t)((kk + 1) * 32 + half * 16) * HWSZ], 0, 1);

    // B fragment: lane = pixel n; lanes 0-15 hold K 0..15, lanes 16-31 K 16..31
    v16bf bfrag;
    #pragma unroll
    for (int j = 0; j < 8; ++j) {
      int k0 = kk * 32 + half * 16 + 2 * j;
      float f0 = fbase[(size_t)k0 * HWSZ];
      float f1 = fbase[(size_t)(k0 + 1) * HWSZ];
      bfrag[2*j]     = (__bf16)f0;
      bfrag[2*j + 1] = (__bf16)f1;
    }

    // A fragments: low lanes K{0..15}, high lanes K{16..31}
    const int koff = half ? 16 : 0;
    v16bf afr[5];
    #pragma unroll
    for (int t = 0; t < 5; ++t) {
      int row = t * 16 + n;
      ((uint4*)&afr[t])[0] = *(const uint4*)(chunk + row * 32 + koff);
      ((uint4*)&afr[t])[1] = *(const uint4*)(chunk + row * 32 + koff + 8);
    }
    #pragma unroll
    for (int t = 0; t < 5; ++t)
      acc[t] = __builtin_amdgcn_wmma_f32_16x16x32_bf16(
          false, afr[t], false, bfrag, (short)0, acc[t], false, false);
  }

  // masked epilogue: D layout: lane n holds pixel n, VGPR r -> proto row r+8*half
  const int m = mask[p];
  const float inv = invn[p];
  float* out = msim + (size_t)p * KPC;
  if (m >= 1 && m <= NCLS) {
    const int lo = (m - 1) * KPC;
    #pragma unroll
    for (int t = 0; t < 5; ++t)
      #pragma unroll
      for (int r = 0; r < 8; ++r) {
        int g = t * 16 + half * 8 + r;
        int k = g - lo;
        if (k >= 0 && k < KPC) out[k] = acc[t][r] * inv;
      }
  } else if (half == 0) {
    #pragma unroll
    for (int k = 0; k < KPC; ++k) out[k] = 0.f;
  }
}

// ---------------------------------------------------------------------------
// K2: 7x7 window pooling -> region rows -> sufficient statistics per (b,c)
//     region = num_sum/(den_sum + 49*EPS); valid <=> den_sum >= 3
//     LDS ds_add_f32 accumulation, one global atomic per value per block.
// ---------------------------------------------------------------------------
__global__ void __launch_bounds__(256) k2_pool_stats(
    const int* __restrict__ mask, const float* __restrict__ msim,
    float* __restrict__ gstats) {
  __shared__ float sst[NCLS * STATS_PER];  // 536 floats
  const int tid = threadIdx.x;
  const int b   = blockIdx.x / 36;
  const int hw  = (blockIdx.x % 36) * 256 + tid;
  const int h   = hw / WW_, w = hw % WW_;
  const int* mrow = mask + b * HWSZ;

  for (int i = tid; i < NCLS * STATS_PER; i += 256) sst[i] = 0.f;
  __syncthreads();

  int mo = mrow[hw];
  if (mo >= 1 && mo <= NCLS) atomicAdd(&sst[(mo - 1) * STATS_PER + 66], 1.f);

  // window class histogram packed in 2x u32 (8-bit counters, max 49)
  unsigned int clo = 0, chi = 0;
  for (int dy = -3; dy <= 3; ++dy) {
    int hy = h + dy; if (hy < 0 || hy >= HH_) continue;
    for (int dx = -3; dx <= 3; ++dx) {
      int wx = w + dx; if (wx < 0 || wx >= WW_) continue;
      int mm = mrow[hy * WW_ + wx];
      if (mm >= 1 && mm <= NCLS) {
        int c = mm - 1;
        unsigned int inc = 1u << (8 * (c & 3));
        if (c < 4) clo += inc; else chi += inc;
      }
    }
  }

  for (int c = 0; c < NCLS; ++c) {
    int den = (int)(((c < 4 ? clo : chi) >> (8 * (c & 3))) & 0xFF);
    if (den < 3) continue;  // den/49 > 0.05 requires >= 3
    float a[KPC];
    #pragma unroll
    for (int k = 0; k < KPC; ++k) a[k] = 0.f;
    for (int dy = -3; dy <= 3; ++dy) {
      int hy = h + dy; if (hy < 0 || hy >= HH_) continue;
      for (int dx = -3; dx <= 3; ++dx) {
        int wx = w + dx; if (wx < 0 || wx >= WW_) continue;
        int q = hy * WW_ + wx;
        if (mrow[q] == c + 1) {
          const float* ms = msim + (size_t)(b * HWSZ + q) * KPC;
          #pragma unroll
          for (int k = 0; k < KPC; ++k) a[k] += ms[k];
        }
      }
    }
    const float dinv = 1.f / ((float)den + 49.f * 1e-8f);
    #pragma unroll
    for (int k = 0; k < KPC; ++k) a[k] *= dinv;
    float* s = &sst[c * STATS_PER];
    #pragma unroll
    for (int k = 0; k < KPC; ++k) atomicAdd(&s[k], a[k]);
    int idx = 10;
    #pragma unroll
    for (int k = 0; k < KPC; ++k)
      #pragma unroll
      for (int l = k; l < KPC; ++l) { atomicAdd(&s[idx], a[k] * a[l]); ++idx; }
    atomicAdd(&s[65], 1.f);
  }
  __syncthreads();
  for (int i = tid; i < NCLS * STATS_PER; i += 256)
    atomicAdd(&gstats[b * NCLS * STATS_PER + i], sst[i]);
}

// ---------------------------------------------------------------------------
// K3: gram from sufficient statistics, off-diagonal loss, active mask, reduce
// ---------------------------------------------------------------------------
__global__ void k3_final(const float* __restrict__ gstats, float* __restrict__ out) {
  __shared__ float s_l[64], s_a[64], s_img[NB];
  int i = threadIdx.x;
  if (i < 64) {
    const float* st = gstats + i * STATS_PER;
    float V = st[65], om = st[66];
    float Vf = fmaxf(V, 1.f);
    float s1[KPC], nrm[KPC];
    #pragma unroll
    for (int k = 0; k < KPC; ++k) s1[k] = st[k];
    {
      int idx = 10;
      #pragma unroll
      for (int k = 0; k < KPC; ++k)
        #pragma unroll
        for (int l = k; l < KPC; ++l) {
          if (l == k) {
            float c = st[idx] - s1[k] * s1[k] / Vf;
            nrm[k] = sqrtf(fmaxf(c, 0.f));
          }
          ++idx;
        }
    }
    float loss = 0.f;
    {
      int idx = 10;
      #pragma unroll
      for (int k = 0; k < KPC; ++k)
        #pragma unroll
        for (int l = k; l < KPC; ++l) {
          if (l > k) {
            float c = st[idx] - s1[k] * s1[l] / Vf;
            float g = c / ((nrm[k] + 1e-8f) * (nrm[l] + 1e-8f) * Vf);
            loss += 2.f * g * g;   // both (k,l) and (l,k) off-diagonals
          }
          ++idx;
        }
    }
    float lc  = loss / (float)(KPC * (KPC - 1));
    float act = (om >= 1.f && V >= 1.5f) ? 1.f : 0.f;
    s_l[i] = lc * act; s_a[i] = act;
  }
  __syncthreads();
  if (i < NB) {
    float la = 0.f, na = 0.f;
    for (int c = 0; c < NCLS; ++c) { la += s_l[i * NCLS + c]; na += s_a[i * NCLS + c]; }
    s_img[i] = la / fmaxf(na, 1.f);
  }
  __syncthreads();
  if (i == 0) {
    float t = 0.f;
    for (int b = 0; b < NB; ++b) t += s_img[b];
    out[0] = t / (float)NB;
  }
}

// ---------------------------------------------------------------------------
extern "C" void kernel_launch(void* const* d_in, const int* in_sizes, int n_in,
                              void* d_out, int out_size, void* d_ws, size_t ws_size,
                              hipStream_t stream) {
  (void)in_sizes; (void)n_in; (void)out_size; (void)ws_size;
  const float* feat  = (const float*)d_in[0];
  const float* proto = (const float*)d_in[1];
  const int*   mask  = (const int*)d_in[2];
  float* out = (float*)d_out;

  char* w = (char*)d_ws;
  float*          invn   = (float*)w;                                  // 294912 B
  unsigned short* pbf    = (unsigned short*)(w + 294912);              //  81920 B
  float*          msim   = (float*)(w + 294912 + 81920);               // 2949120 B
  float*          gstats = (float*)(w + 294912 + 81920 + 2949120);     //  17152 B

  k0_inv_norm<<<NPIX / 256, 256, 0, stream>>>(feat, invn);
  kp_proto<<<NPROTO, 64, 0, stream>>>(proto, pbf);
  kzero<<<(NB * NCLS * STATS_PER + 255) / 256, 256, 0, stream>>>(gstats, NB * NCLS * STATS_PER);
  k1_wmma_sim<<<NPIX / 128, 256, 0, stream>>>(feat, pbf, mask, invn, msim);
  k2_pool_stats<<<NB * 36, 256, 0, stream>>>(mask, msim, gstats);
  k3_final<<<1, 64, 0, stream>>>(gstats, out);
}
